// model_2508260900997
// MI455X (gfx1250) — matrix operand order repaired and checked by reading, not compiled, not run on hardware
//
#include <hip/hip_runtime.h>
#include <hip/hip_bf16.h>
#include <cstdint>

typedef unsigned short u16;
typedef __attribute__((ext_vector_type(16))) __bf16 v16bf;
typedef __attribute__((ext_vector_type(8)))  float  v8f;

#define HW    16384
#define NPAT  961
#define NPATP 992

__device__ __forceinline__ u16 f2bf(float f) {
  unsigned u = __float_as_uint(f);
  u += 0x7FFFu + ((u >> 16) & 1u);
  return (u16)(u >> 16);
}

// =============================== WMMA GEMM ===============================
// C[M,N] = A[M,K] * B, A row-major bf16 (lda%8==0), B either (K,N) [BNK=0]
// or (N,K) [BNK=1] row-major bf16. K (=Kpad) must be a multiple of 32 and
// zero-padded; KN-layout B must have ldb >= roundup(N,256) (%8==0).
// Block tile 64x256, 8 waves (2x4), per-wave tile 32x64 -> 8 WMMA / K-step.
// Clamped loads + bounds-checked stores handle ragged M/N. Epilogue: bias
// (none/per-m/per-n), activation (none/lrelu/sigmoid(lrelu)), dual f32+bf16.
union FragU { v16bf v; uint4 q[2]; };

template <int BNK>
__global__ __launch_bounds__(256)
void gemm_bf16_wmma(const u16* __restrict__ A, int lda, int M,
                    const u16* __restrict__ B, int ldb, int N, int Kpad,
                    float* __restrict__ Cf, int ldc,
                    u16* __restrict__ Cb, int ldcb,
                    const float* __restrict__ bias, int biasMode, int actMode)
{
  __shared__ __align__(16) u16 Al[64][48];    // [m][k], padded stride
  __shared__ __align__(16) u16 Bl[256][48];   // [n][k]
  const int tid  = threadIdx.x;
  const int m0   = blockIdx.y * 64;
  const int n0   = blockIdx.x * 256;
  const int wave = tid >> 5, lane = tid & 31;
  const int wm   = wave >> 2, wn = wave & 3;  // 2x4 waves
  const int half = lane >> 4, l16 = lane & 15;

  v8f acc[2][4];
#pragma unroll
  for (int i = 0; i < 2; ++i)
#pragma unroll
    for (int j = 0; j < 4; ++j)
#pragma unroll
      for (int r = 0; r < 8; ++r) acc[i][j][r] = 0.0f;

  // A staging: 64 rows x 32 cols, one uint4 (8 bf16) per thread
  const int am = tid >> 2;
  const int ak = (tid & 3) << 3;
  int arow = m0 + am; if (arow >= M) arow = M - 1;
  const u16* aptr = A + (size_t)arow * lda + ak;

  int bn_, bk_;
  const u16* bptr;
  if (BNK) {            // B (N,K): 256 rows x 32 cols, 4 uint4 per thread
    bn_ = tid;
    bk_ = 0;
    int brow = n0 + bn_; if (brow >= N) brow = N - 1;
    bptr = B + (size_t)brow * ldb;
  } else {              // B (K,N): read 32 along n, transpose into Bl[n][k]
    bk_ = tid >> 3;           // 0..31
    bn_ = (tid & 7) << 5;     // 0..224
    bptr = B + (size_t)bk_ * ldb + n0 + bn_;
  }

  for (int k = 0; k < Kpad; k += 32) {
    uint4 av = *(const uint4*)(aptr + k);
    uint4 bv0, bv1, bv2, bv3;
    if (BNK) {
      const uint4* p = (const uint4*)(bptr + k);
      bv0 = p[0]; bv1 = p[1]; bv2 = p[2]; bv3 = p[3];
    } else {
      const uint4* p = (const uint4*)(bptr + (size_t)k * ldb);
      bv0 = p[0]; bv1 = p[1]; bv2 = p[2]; bv3 = p[3];
    }
    __syncthreads();
    *(uint4*)&Al[am][ak] = av;
    if (BNK) {
      *(uint4*)&Bl[bn_][0]  = bv0;
      *(uint4*)&Bl[bn_][8]  = bv1;
      *(uint4*)&Bl[bn_][16] = bv2;
      *(uint4*)&Bl[bn_][24] = bv3;
    } else {
      u16 tmp[32];
      *(uint4*)&tmp[0]  = bv0;
      *(uint4*)&tmp[8]  = bv1;
      *(uint4*)&tmp[16] = bv2;
      *(uint4*)&tmp[24] = bv3;
#pragma unroll
      for (int j = 0; j < 32; ++j) Bl[bn_ + j][bk_] = tmp[j];
    }
    __syncthreads();

    // Fragment loads per ISA 16-bit VGPR layouts (wave32):
    // A: lanes0-15 m=l, elems0-7:k=half*8+i, elems8-15:k=16+half*8+(i-8)
    // B: lane n=l16, elems0-7:k=half*8+i, elems8-15:k=16+half*8+(i-8)
    FragU a[2], b[4];
#pragma unroll
    for (int fm = 0; fm < 2; ++fm) {
      a[fm].q[0] = *(const uint4*)&Al[wm * 32 + fm * 16 + l16][half * 8];
      a[fm].q[1] = *(const uint4*)&Al[wm * 32 + fm * 16 + l16][16 + half * 8];
    }
#pragma unroll
    for (int fn = 0; fn < 4; ++fn) {
      b[fn].q[0] = *(const uint4*)&Bl[wn * 64 + fn * 16 + l16][half * 8];
      b[fn].q[1] = *(const uint4*)&Bl[wn * 64 + fn * 16 + l16][16 + half * 8];
    }
#pragma unroll
    for (int fm = 0; fm < 2; ++fm)
#pragma unroll
      for (int fn = 0; fn < 4; ++fn)
        acc[fm][fn] = __builtin_amdgcn_wmma_f32_16x16x32_bf16(
            false, a[fm].v, false, b[fn].v, (short)0, acc[fm][fn], false, false);
  }

#pragma unroll
  for (int fm = 0; fm < 2; ++fm)
#pragma unroll
    for (int fn = 0; fn < 4; ++fn) {
      const int nn = n0 + wn * 64 + fn * 16 + l16;
#pragma unroll
      for (int r = 0; r < 8; ++r) {
        const int mm = m0 + wm * 32 + fm * 16 + half * 8 + r;
        if (mm < M && nn < N) {
          float v = acc[fm][fn][r];
          if (biasMode == 1) v += bias[mm];
          else if (biasMode == 2) v += bias[nn];
          if (actMode >= 1) v = v > 0.f ? v : 0.2f * v;
          if (actMode == 2) v = 1.f / (1.f + __expf(-v));
          if (Cf) Cf[(size_t)mm * ldc + nn] = v;
          if (Cb) Cb[(size_t)mm * ldcb + nn] = f2bf(v);
        }
      }
    }
}

// =========================== support kernels ============================
__global__ void k_bilinear4x(const float* __restrict__ src, float* __restrict__ dst, int total) {
  int i = blockIdx.x * 256 + threadIdx.x;
  if (i >= total) return;
  int w = i & 127, h = (i >> 7) & 127, bc = i >> 14;
  float sy = h * 0.25f - 0.375f, sx = w * 0.25f - 0.375f;
  int y0 = (int)floorf(sy), x0 = (int)floorf(sx);
  float fy = sy - (float)y0, fx = sx - (float)x0;
  int y1 = y0 + 1, x1 = x0 + 1;
  y0 = min(max(y0, 0), 31); y1 = min(max(y1, 0), 31);
  x0 = min(max(x0, 0), 31); x1 = min(max(x1, 0), 31);
  const float* s = src + (size_t)bc * 1024;
  float v00 = s[y0 * 32 + x0], v01 = s[y0 * 32 + x1];
  float v10 = s[y1 * 32 + x0], v11 = s[y1 * 32 + x1];
  dst[i] = (v00 * (1.f - fx) + v01 * fx) * (1.f - fy) + (v10 * (1.f - fx) + v11 * fx) * fy;
}

// reflect-pad 3x3 im2col: col[(c*9+j)][p] bf16, rows >= Cin*9 zeroed to Kpad
__global__ void k_im2col_r3(const float* __restrict__ x, u16* __restrict__ col, int Cin, int Kpad) {
  long long i = (long long)blockIdx.x * 256 + threadIdx.x;
  if (i >= (long long)Kpad * HW) return;
  int p = (int)(i & (HW - 1));
  int rr = (int)(i >> 14);
  float v = 0.f;
  if (rr < Cin * 9) {
    int c = rr / 9, j = rr - c * 9;
    int h = (p >> 7) + j / 3 - 1, w = (p & 127) + j % 3 - 1;
    h = h < 0 ? -h : (h > 127 ? 254 - h : h);
    w = w < 0 ? -w : (w > 127 ? 254 - w : w);
    v = x[(size_t)c * HW + (h << 7) + w];
  }
  col[i] = f2bf(v);
}

// patches (K=8,S=4): P[p][c*64+kr*8+kc] = x[c][r0*4+kr][c0*4+kc], pad rows zero
__global__ void k_patches(const float* __restrict__ x, u16* __restrict__ P, int prows, int ldp) {
  long long i = (long long)blockIdx.x * 256 + threadIdx.x;
  if (i >= (long long)prows * ldp) return;
  int f = (int)(i % ldp);
  int p = (int)(i / ldp);
  float v = 0.f;
  if (p < NPAT) {
    int r0 = p / 31, c0 = p - r0 * 31;
    int c = f >> 6, kk = f & 63;
    v = x[(size_t)c * HW + ((r0 * 4 + (kk >> 3)) << 7) + (c0 * 4 + (kk & 7))];
  }
  P[i] = f2bf(v);
}

// fold: gather of overlapping patch contributions
__global__ void k_fold(const float* __restrict__ Y, float* __restrict__ out, int C, int ldn) {
  long long i = (long long)blockIdx.x * 256 + threadIdx.x;
  if (i >= (long long)C * HW) return;
  int w = (int)(i & 127), h = (int)((i >> 7) & 127), c = (int)(i >> 14);
  int rlo = h >= 7 ? (h - 4) >> 2 : 0, rhi = min(30, h >> 2);
  int clo = w >= 7 ? (w - 4) >> 2 : 0, chi = min(30, w >> 2);
  float s = 0.f;
  for (int r0 = rlo; r0 <= rhi; ++r0)
    for (int c0 = clo; c0 <= chi; ++c0)
      s += Y[(size_t)(r0 * 31 + c0) * ldn + c * 64 + (h - 4 * r0) * 8 + (w - 4 * c0)];
  out[i] = s;
}

// masked softmax row kernel: A = softmax(score*sigmoid(Ti)*scale) * (mask!=0)
__global__ void k_softmax_mask(const float* __restrict__ score, int lds_,
                               const float* __restrict__ Ti, int ldt,
                               u16* __restrict__ Abf, int lda, int N, float scale)
{
  __shared__ float sv[1024];
  __shared__ unsigned char mz[1024];
  __shared__ float red[256];
  int row = blockIdx.x, tid = threadIdx.x;
  const float* sr = score + (size_t)row * lds_;
  const float* tr = Ti + (size_t)row * ldt;
  float mx = -3.0e38f;
  for (int j = tid; j < N; j += 256) {
    float m = 1.f / (1.f + __expf(-tr[j]));
    float s = sr[j] * m * scale;
    sv[j] = s; mz[j] = (m == 0.f);
    mx = fmaxf(mx, s);
  }
  red[tid] = mx; __syncthreads();
  for (int k = 128; k > 0; k >>= 1) { if (tid < k) red[tid] = fmaxf(red[tid], red[tid + k]); __syncthreads(); }
  mx = red[0]; __syncthreads();
  float sum = 0.f;
  for (int j = tid; j < N; j += 256) { float e = __expf(sv[j] - mx); sv[j] = e; sum += e; }
  red[tid] = sum; __syncthreads();
  for (int k = 128; k > 0; k >>= 1) { if (tid < k) red[tid] += red[tid + k]; __syncthreads(); }
  float inv = 1.f / red[0];
  u16* ar = Abf + (size_t)row * lda;
  for (int j = tid; j < lda; j += 256)
    ar[j] = (j < N && !mz[j]) ? f2bf(sv[j] * inv) : (u16)0;
}

__global__ void k_rowmean(const float* __restrict__ x, float* __restrict__ mean, int cols) {
  __shared__ float red[256];
  int row = blockIdx.x, tid = threadIdx.x;
  const float* r = x + (size_t)row * cols;
  float s = 0.f;
  for (int j = tid; j < cols; j += 256) s += r[j];
  red[tid] = s; __syncthreads();
  for (int k = 128; k > 0; k >>= 1) { if (tid < k) red[tid] += red[tid + k]; __syncthreads(); }
  if (tid == 0) mean[row] = red[0] / (float)cols;
}

__global__ void k_matvec256(const float* __restrict__ W, const float* __restrict__ b,
                            const float* __restrict__ v, float* __restrict__ out) {
  __shared__ float vs[256];
  int tid = threadIdx.x;
  vs[tid] = v[tid]; __syncthreads();
  float s = b[tid];
  for (int j = 0; j < 256; ++j) s += W[tid * 256 + j] * vs[j];
  out[tid] = s;
}

__global__ void k_outer256(const float* __restrict__ a, const float* __restrict__ b,
                           float* __restrict__ sf, u16* __restrict__ sb) {
  int i = blockIdx.x * 256 + threadIdx.x;
  float v = a[i >> 8] * b[i & 255];
  sf[i] = v; sb[i] = f2bf(v);
}

__global__ void k_pack(const float* __restrict__ src, u16* __restrict__ dst,
                       int srows, int scols, int drows, int dcols) {
  long long i = (long long)blockIdx.x * 256 + threadIdx.x;
  if (i >= (long long)drows * dcols) return;
  int r = (int)(i / dcols), c = (int)(i % dcols);
  float v = (r < srows && c < scols) ? src[(size_t)r * scols + c] : 0.f;
  dst[i] = f2bf(v);
}

__global__ void k_packsum(const float* __restrict__ a, const float* __restrict__ b, u16* __restrict__ dst,
                          int srows, int scols, int drows, int dcols) {
  long long i = (long long)blockIdx.x * 256 + threadIdx.x;
  if (i >= (long long)drows * dcols) return;
  int r = (int)(i / dcols), c = (int)(i % dcols);
  float v = (r < srows && c < scols) ? a[(size_t)r * scols + c] + b[(size_t)r * scols + c] : 0.f;
  dst[i] = f2bf(v);
}

__global__ void k_addvec(const float* a, const float* b, float* o, int n) {
  int i = blockIdx.x * 256 + threadIdx.x; if (i < n) o[i] = a[i] + b[i];
}
__global__ void k_zero_us(u16* p, long long n) {
  long long i = (long long)blockIdx.x * 256 + threadIdx.x; if (i < n) p[i] = 0;
}
__global__ void k_addinp(float* d, const float* a, long long n) {
  long long i = (long long)blockIdx.x * 256 + threadIdx.x; if (i < n) d[i] += a[i];
}
__global__ void k_add2(float* o, const float* a, const float* b, long long n) {
  long long i = (long long)blockIdx.x * 256 + threadIdx.x; if (i < n) o[i] = a[i] + b[i];
}
__global__ void k_halfsum_bf(const float* a, const float* b, u16* o, long long n) {
  long long i = (long long)blockIdx.x * 256 + threadIdx.x; if (i < n) o[i] = f2bf((a[i] + b[i]) * 0.5f);
}
__global__ void k_iaff_comb(const float* xs, const float* xe, const float* w,
                            const float* f, float* o, long long n) {
  long long i = (long long)blockIdx.x * 256 + threadIdx.x;
  if (i < n) { float wi = w[i]; o[i] = xs[i] * wi + xe[i] * (1.f - wi) + f[i]; }
}

// ============================ orchestration =============================
extern "C" void kernel_launch(void* const* d_in, const int* in_sizes, int n_in,
                              void* d_out, int out_size, void* d_ws, size_t ws_size,
                              hipStream_t stream)
{
  (void)in_sizes; (void)n_in; (void)out_size; (void)ws_size;
  const float* hs      = (const float*)d_in[0];   // (4,191,32,32)
  const float* ms      = (const float*)d_in[1];   // (4,6,128,128)
  const float* w_ms_c1 = (const float*)d_in[2];   // (3,6,3,3)
  const float* w_ms_c2 = (const float*)d_in[3];   // (32,3,1,1)
  const float* w_cv_c1 = (const float*)d_in[4];   // (98,197,3,3)
  const float* w_cv_c2 = (const float*)d_in[5];   // (256,98,1,1)
  const float* w_out   = (const float*)d_in[6];   // (191,256,3,3)
#define SPA(i, k) ((const float*)d_in[7  + (i) * 14 + (k)])
#define SPE(i, k) ((const float*)d_in[49 + (i) * 12 + (k)])
#define IAF(i, k) ((const float*)d_in[85 + (i) * 4  + (k)])
  float* dout = (float*)d_out;

  char* wsb = (char*)d_ws;
  size_t off = 0;
  auto alloc = [&](size_t bytes) -> char* {
    char* p = wsb + off;
    off = (off + bytes + 255) & ~(size_t)255;
    return p;
  };

  // persistent
  float* hs_up = (float*)alloc(4UL * 191 * HW * 4);
  float* xbuf  = (float*)alloc(4UL * 256 * HW * 4);
  float* msf   = (float*)alloc(4UL * 32 * HW * 4);
  u16*   Aspa  = (u16*)alloc(4UL * NPAT * NPATP * 2);
  u16*   Aspe  = (u16*)alloc(4UL * 256 * 256 * 2);
  // packed bf16 weights
  u16* msc1b = (u16*)alloc(3UL * 64 * 2);
  u16* msc2b = (u16*)alloc(32UL * 32 * 2);
  u16* cvc1b = (u16*)alloc(98UL * 1792 * 2);
  u16* cvc2b = (u16*)alloc(256UL * 128 * 2);
  u16* outwb = (u16*)alloc(191UL * 2304 * 2);
  u16 *spc1b[3], *spc2b[3], *spgb[3], *spWb[3], *spthb[3], *spfcb[3], *spMs[3];
  u16 *sec1b[3], *sec2b[3], *seedb[3], *seowb[3], *seMs[3];
  u16 *iac1b[3], *iac2b[3], *ialab[3];
  float *spbs[3], *sebs[3];
  for (int L = 0; L < 3; ++L) {
    spc1b[L] = (u16*)alloc(16UL * 288 * 2);   spc2b[L] = (u16*)alloc(32UL * 32 * 2);
    spgb[L]  = (u16*)alloc(16UL * 288 * 2);   spWb[L]  = (u16*)alloc(16UL * 288 * 2);
    spthb[L] = (u16*)alloc(32UL * 32 * 2);    spfcb[L] = (u16*)alloc(256UL * 1024 * 2);
    spMs[L]  = (u16*)alloc((size_t)NPAT * NPATP * 2);
    spbs[L]  = (float*)alloc(NPAT * 4);
    sec1b[L] = (u16*)alloc(128UL * 2304 * 2); sec2b[L] = (u16*)alloc(256UL * 128 * 2);
    seedb[L] = (u16*)alloc(256UL * 256 * 2);  seowb[L] = (u16*)alloc(256UL * 2304 * 2);
    seMs[L]  = (u16*)alloc(256UL * 256 * 2);  sebs[L]  = (float*)alloc(256 * 4);
    iac1b[L] = (u16*)alloc(128UL * 2304 * 2); iac2b[L] = (u16*)alloc(256UL * 128 * 2);
    ialab[L] = (u16*)alloc(256UL * 256 * 2);
  }
  // scratch
  u16*   col   = (u16*)alloc(2304UL * HW * 2);
  float* xc    = (float*)alloc(197UL * HW * 4);
  float* t1    = (float*)alloc(256UL * HW * 4);
  float* t2    = (float*)alloc(256UL * HW * 4);
  float* t3    = (float*)alloc(256UL * HW * 4);
  float* t4    = (float*)alloc(256UL * HW * 4);
  float* tbig  = (float*)alloc((size_t)NPAT * HW * 4);
  u16*   tb1   = (u16*)alloc(256UL * HW * 2);
  u16*   tb2   = (u16*)alloc(256UL * HW * 2);
  u16*   tb3   = (u16*)alloc(256UL * HW * 2);
  u16*   x1bf  = (u16*)alloc(32UL * HW * 2);
  float* x1f   = (float*)alloc(16UL * HW * 4);
  float* x2f   = (float*)alloc(16UL * HW * 4);
  float* x3f   = (float*)alloc(32UL * HW * 4);
  u16*   P1    = (u16*)alloc((size_t)NPAT * 1024 * 2);
  u16*   P2    = (u16*)alloc((size_t)NPAT * 1024 * 2);
  u16*   P3    = (u16*)alloc((size_t)NPATP * 2048 * 2);
  u16*   Pf    = (u16*)alloc((size_t)NPATP * HW * 2);
  u16*   wib   = (u16*)alloc((size_t)NPAT * 256 * 2);
  u16*   xib   = (u16*)alloc((size_t)NPAT * 256 * 2);
  float* scoref  = (float*)alloc((size_t)NPAT * NPAT * 4);
  u16*   scorebf = (u16*)alloc((size_t)NPATP * 1024 * 2);
  float* Tif     = (float*)alloc((size_t)NPAT * NPAT * 4);
  float* yif     = (float*)alloc((size_t)NPAT * 2048 * 4);
  float* zf      = (float*)alloc(32UL * HW * 4);
  float* xmean   = (float*)alloc(256 * 4);
  float* mbn     = (float*)alloc(256 * 4);
  float* mbe     = (float*)alloc(256 * 4);
  float* sc256f  = (float*)alloc(256UL * 256 * 4);
  u16*   sc256b  = (u16*)alloc(256UL * 256 * 2);
  float* Ti256   = (float*)alloc(256UL * 256 * 4);

  auto G1 = [&](long long total) { return dim3((unsigned)((total + 255) / 256)); };
  auto gemm = [&](const u16* A, int lda, int M, const u16* B, int ldb, int N, int K, int bnk,
                  float* Cf, int ldc, u16* Cb, int ldcb, const float* bias, int bm, int act) {
    dim3 g((N + 255) / 256, (M + 63) / 64);
    if (bnk) gemm_bf16_wmma<1><<<g, 256, 0, stream>>>(A, lda, M, B, ldb, N, K, Cf, ldc, Cb, ldcb, bias, bm, act);
    else     gemm_bf16_wmma<0><<<g, 256, 0, stream>>>(A, lda, M, B, ldb, N, K, Cf, ldc, Cb, ldcb, bias, bm, act);
  };
  auto pack = [&](const float* s, u16* d, int sr, int sc, int dr, int dc) {
    k_pack<<<G1((long long)dr * dc), 256, 0, stream>>>(s, d, sr, sc, dr, dc);
  };

  // ---------- weight packing (bf16, K zero-padded to mult of 32) ----------
  pack(w_ms_c1, msc1b, 3, 54, 3, 64);
  pack(w_ms_c2, msc2b, 32, 3, 32, 32);
  pack(w_cv_c1, cvc1b, 98, 1773, 98, 1792);
  pack(w_cv_c2, cvc2b, 256, 98, 256, 128);
  pack(w_out,   outwb, 191, 2304, 191, 2304);
  for (int L = 0; L < 3; ++L) {
    pack(SPA(L, 0), spc1b[L], 16, 288, 16, 288);
    pack(SPA(L, 1), spc2b[L], 32, 16, 32, 32);
    pack(SPA(L, 2), spgb[L],  16, 288, 16, 288);
    pack(SPA(L, 4), spWb[L],  16, 288, 16, 288);
    pack(SPA(L, 6), spthb[L], 32, 16, 32, 32);
    pack(SPA(L, 8), spfcb[L], 256, 1024, 256, 1024);
    k_packsum<<<G1((long long)NPAT * NPATP), 256, 0, stream>>>(SPA(L, 10), SPA(L, 12), spMs[L], NPAT, NPAT, NPAT, NPATP);
    k_addvec<<<G1(NPAT), 256, 0, stream>>>(SPA(L, 11), SPA(L, 13), spbs[L], NPAT);
    pack(SPE(L, 0), sec1b[L], 128, 2304, 128, 2304);
    pack(SPE(L, 1), sec2b[L], 256, 128, 256, 128);
    pack(SPE(L, 4), seedb[L], 256, 256, 256, 256);
    pack(SPE(L, 6), seowb[L], 256, 2304, 256, 2304);
    k_packsum<<<G1(256LL * 256), 256, 0, stream>>>(SPE(L, 8), SPE(L, 10), seMs[L], 256, 256, 256, 256);
    k_addvec<<<G1(256), 256, 0, stream>>>(SPE(L, 9), SPE(L, 11), sebs[L], 256);
    pack(IAF(L, 0), iac1b[L], 128, 2304, 128, 2304);
    pack(IAF(L, 1), iac2b[L], 256, 128, 256, 128);
    pack(IAF(L, 2), ialab[L], 256, 256, 256, 256);
  }

  // ------------------------------ stage 0 ------------------------------
  k_bilinear4x<<<G1(4LL * 191 * HW), 256, 0, stream>>>(hs, hs_up, 4 * 191 * HW);
  for (int b = 0; b < 4; ++b) {
    const float* ms_b = ms + (size_t)b * 6 * HW;
    // ms_f = basic_block(ms, ms_conv)
    k_im2col_r3<<<G1(64LL * HW), 256, 0, stream>>>(ms_b, col, 6, 64);
    k_zero_us<<<G1(29LL * HW), 256, 0, stream>>>(x1bf + 3 * HW, 29LL * HW);
    gemm(msc1b, 64, 3, col, HW, HW, 64, 0, nullptr, 0, x1bf, HW, nullptr, 0, 1);
    gemm(msc2b, 32, 32, x1bf, HW, HW, 32, 0, msf + (size_t)b * 32 * HW, HW, nullptr, 0, nullptr, 0, 1);
    // x = basic_block(concat(hs_up, ms), conv)
    hipMemcpyAsync(xc, hs_up + (size_t)b * 191 * HW, (size_t)191 * HW * 4, hipMemcpyDeviceToDevice, stream);
    hipMemcpyAsync(xc + (size_t)191 * HW, ms_b, (size_t)6 * HW * 4, hipMemcpyDeviceToDevice, stream);
    k_im2col_r3<<<G1(1792LL * HW), 256, 0, stream>>>(xc, col, 197, 1792);
    k_zero_us<<<G1(30LL * HW), 256, 0, stream>>>(tb2 + 98 * HW, 30LL * HW);
    gemm(cvc1b, 1792, 98, col, HW, HW, 1792, 0, nullptr, 0, tb2, HW, nullptr, 0, 1);
    gemm(cvc2b, 128, 256, tb2, HW, HW, 128, 0, xbuf + (size_t)b * 256 * HW, HW, nullptr, 0, nullptr, 0, 1);
  }

  // ------------------------------- layers -------------------------------
  for (int L = 0; L < 3; ++L) {
    // ---- spa ----
    for (int b = 0; b < 4; ++b) {
      float* mf = msf + (size_t)b * 32 * HW;
      k_im2col_r3<<<G1(288LL * HW), 256, 0, stream>>>(mf, col, 32, 288);
      k_zero_us<<<G1(16LL * HW), 256, 0, stream>>>(x1bf + 16 * HW, 16LL * HW);
      gemm(spgb[L], 288, 16, col, HW, HW, 288, 0, x1f, HW, x1bf, HW, SPA(L, 3), 1, 0);
      gemm(spWb[L], 288, 16, col, HW, HW, 288, 0, x2f, HW, nullptr, 0, SPA(L, 5), 1, 0);
      gemm(spthb[L], 32, 32, x1bf, HW, HW, 32, 0, x3f, HW, nullptr, 0, SPA(L, 7), 1, 0);
      k_patches<<<G1((long long)NPAT * 1024), 256, 0, stream>>>(x2f, P2, NPAT, 1024);
      k_patches<<<G1((long long)NPAT * 1024), 256, 0, stream>>>(x1f, P1, NPAT, 1024);
      k_patches<<<G1((long long)NPATP * 2048), 256, 0, stream>>>(x3f, P3, NPATP, 2048);
      gemm(P2, 1024, NPAT, spfcb[L], 1024, 256, 1024, 1, nullptr, 0, wib, 256, SPA(L, 9), 2, 0);
      gemm(P1, 1024, NPAT, spfcb[L], 1024, 256, 1024, 1, nullptr, 0, xib, 256, SPA(L, 9), 2, 0);
      k_zero_us<<<G1(31LL * 1024), 256, 0, stream>>>(scorebf + (size_t)NPAT * 1024, 31LL * 1024);
      gemm(wib, 256, NPAT, xib, 256, NPAT, 256, 1, scoref, NPAT, scorebf, 1024, nullptr, 0, 0);
      gemm(spMs[L], NPATP, NPAT, scorebf, 1024, NPAT, NPATP, 0, Tif, NPAT, nullptr, 0, spbs[L], 1, 0);
      u16* Ab = Aspa + (size_t)b * NPAT * NPATP;
      k_softmax_mask<<<NPAT, 256, 0, stream>>>(scoref, NPAT, Tif, NPAT, Ab, NPATP, NPAT, 0.0625f);
      gemm(Ab, NPATP, NPAT, P3, 2048, 2048, NPATP, 0, yif, 2048, nullptr, 0, nullptr, 0, 0);
      k_fold<<<G1(32LL * HW), 256, 0, stream>>>(yif, zf, 32, 2048);
      // ms_f += basic_block(z)
      k_im2col_r3<<<G1(288LL * HW), 256, 0, stream>>>(zf, col, 32, 288);
      k_zero_us<<<G1(16LL * HW), 256, 0, stream>>>(x1bf + 16 * HW, 16LL * HW);
      gemm(spc1b[L], 288, 16, col, HW, HW, 288, 0, nullptr, 0, x1bf, HW, nullptr, 0, 1);
      gemm(spc2b[L], 32, 32, x1bf, HW, HW, 32, 0, t2, HW, nullptr, 0, nullptr, 0, 1);
      k_addinp<<<G1(32LL * HW), 256, 0, stream>>>(mf, t2, 32LL * HW);
    }
    // ---- spe ----
    for (int b = 0; b < 4; ++b) {
      float* xb = xbuf + (size_t)b * 256 * HW;
      k_im2col_r3<<<G1(2304LL * HW), 256, 0, stream>>>(xb, col, 256, 2304);
      gemm(sec1b[L], 2304, 128, col, HW, HW, 2304, 0, nullptr, 0, tb2, HW, nullptr, 0, 1);
      gemm(sec2b[L], 128, 256, tb2, HW, HW, 128, 0, xb, HW, tb1, HW, nullptr, 0, 1);
      // bn only feeds a mean: mean(node_w@x+b) = node_w@mean(x)+b
      k_rowmean<<<256, 256, 0, stream>>>(xb, xmean, HW);
      k_matvec256<<<1, 256, 0, stream>>>(SPE(L, 2), SPE(L, 3), xmean, mbn);
      k_matvec256<<<1, 256, 0, stream>>>(SPE(L, 4), SPE(L, 5), xmean, mbe);
      gemm(seedb[L], 256, 256, tb1, HW, HW, 256, 0, nullptr, 0, tb3, HW, SPE(L, 5), 1, 0); // be
      k_outer256<<<256, 256, 0, stream>>>(mbn, mbe, sc256f, sc256b);
      gemm(seMs[L], 256, 256, sc256b, 256, 256, 256, 0, Ti256, 256, nullptr, 0, sebs[L], 1, 0);
      u16* Ae = Aspe + (size_t)b * 256 * 256;
      k_softmax_mask<<<256, 256, 0, stream>>>(sc256f, 256, Ti256, 256, Ae, 256, 256, 0.0078125f);
      gemm(Ae, 256, 256, tb3, HW, HW, 256, 0, t2, HW, nullptr, 0, nullptr, 0, 0);       // y = A@be
      k_im2col_r3<<<G1(2304LL * HW), 256, 0, stream>>>(t2, col, 256, 2304);
      gemm(seowb[L], 2304, 256, col, HW, HW, 2304, 0, t3, HW, nullptr, 0, SPE(L, 7), 1, 0);
      k_addinp<<<G1(256LL * HW), 256, 0, stream>>>(xb, t3, 256LL * HW);
    }
    // ---- iaff ----
    for (int b = 0; b < 4; ++b) {
      float* xb = xbuf + (size_t)b * 256 * HW;
      k_im2col_r3<<<G1(2304LL * HW), 256, 0, stream>>>(xb, col, 256, 2304);
      gemm(iac1b[L], 2304, 128, col, HW, HW, 2304, 0, nullptr, 0, tb2, HW, nullptr, 0, 1);
      gemm(iac2b[L], 128, 256, tb2, HW, HW, 128, 0, t1, HW, tb1, HW, nullptr, 0, 1);    // fea
      k_patches<<<G1((long long)NPATP * HW), 256, 0, stream>>>(t1, Pf, NPATP, HW);
      gemm(Aspa + (size_t)b * NPAT * NPATP, NPATP, NPAT, Pf, HW, HW, NPATP, 0, tbig, HW, nullptr, 0, nullptr, 0, 0);
      k_fold<<<G1(256LL * HW), 256, 0, stream>>>(tbig, t3, 256, HW);                    // out_spa
      gemm(Aspe + (size_t)b * 256 * 256, 256, 256, tb1, HW, HW, 256, 0, t4, HW, nullptr, 0, nullptr, 0, 0); // out_spe
      k_halfsum_bf<<<G1(256LL * HW), 256, 0, stream>>>(t3, t4, tb3, 256LL * HW);
      gemm(ialab[L], 256, 256, tb3, HW, HW, 256, 0, t2, HW, nullptr, 0, IAF(L, 3), 1, 2); // wei
      k_iaff_comb<<<G1(256LL * HW), 256, 0, stream>>>(t3, t4, t2, t1, xb, 256LL * HW);
    }
  }

  // ------------------------------ output ------------------------------
  for (int b = 0; b < 4; ++b) {
    float* xb = xbuf + (size_t)b * 256 * HW;
    k_im2col_r3<<<G1(2304LL * HW), 256, 0, stream>>>(xb, col, 256, 2304);
    gemm(outwb, 2304, 191, col, HW, HW, 2304, 0, t1, HW, nullptr, 0, nullptr, 0, 1);
    k_add2<<<G1(191LL * HW), 256, 0, stream>>>(dout + (size_t)b * 191 * HW, t1,
                                               hs_up + (size_t)b * 191 * HW, 191LL * HW);
  }
}
